// MultiheadAttention_34007551050213
// MI455X (gfx1250) — matrix operand order repaired and checked by reading, not compiled, not run on hardware
//
#include <hip/hip_runtime.h>
#include <hip/hip_bf16.h>
#include <math.h>

// ---------------------------------------------------------------------------
// Problem constants (reference: N_TOK=2048, TOKDIM=1024, NHEADS=16, KDIM=64)
// ---------------------------------------------------------------------------
#define NTOK 2048
#define DTOK 1024
#define NH   16
#define ZD   64
#define HZ   (NH * ZD)       // 1024
#define BETA 0.125f          // 1/sqrt(64)

typedef __attribute__((ext_vector_type(16))) _Float16 v16h;
typedef __attribute__((ext_vector_type(8)))  float    v8f;

__device__ __forceinline__ v8f zero8() { v8f c = {}; return c; }

__device__ __forceinline__ v8f wmma16(v16h a, v16h b, v8f c) {
  // D = A(16x32,f16) * B(32x16,f16) + C(16x16,f32)
  return __builtin_amdgcn_wmma_f32_16x16x32_f16(
      /*neg_a=*/false, a, /*neg_b=*/false, b,
      /*c_mod=*/(short)0, c, /*reuse_a=*/false, /*reuse_b=*/false);
}

// A-matrix (16x32) from row-major f16: element(row,k) = src[row*ld + k].
// Layout: lane<16 row=lane k={0..7,16..23}; lane>=16 same row k={8..15,24..31}
// -> two contiguous 8xf16 (16B) groups per lane: vectorizes to b128 loads.
__device__ __forceinline__ v16h load_A_f16(const _Float16* src, int ld) {
  const int lane = threadIdx.x & 31;
  const int row  = lane & 15;
  const int kh   = lane >> 4;
  v16h a;
#pragma unroll
  for (int j = 0; j < 16; ++j) {
    int k = j + (j & 8) + 8 * kh;
    a[j] = src[row * ld + k];
  }
  return a;
}

// B-matrix (32x16) from f16: element(k,n) = src[k*ldK + n*ldN].
// Layout: lane = column n (mod 16); lane<16 K={0..7,16..23}, lane>=16 K={8..15,24..31}.
// With ldK==1 each lane reads two contiguous 8xf16 (16B) groups -> b128 loads.
__device__ __forceinline__ v16h load_B_f16(const _Float16* src, int ldK,
                                           int ldN) {
  const int lane = threadIdx.x & 31;
  const int n    = lane & 15;
  const int kh   = lane >> 4;
  v16h b;
#pragma unroll
  for (int j = 0; j < 16; ++j) {
    int k = j + (j & 8) + 8 * kh;
    b[j] = src[k * ldK + n * ldN];
  }
  return b;
}

// A-matrix (16x32, f16) from LDS f16 tile [16][32]
__device__ __forceinline__ v16h load_A_lds(const _Float16* s) {
  const int lane = threadIdx.x & 31;
  const int row  = lane & 15;
  const int kh   = lane >> 4;
  v16h a;
#pragma unroll
  for (int j = 0; j < 16; ++j) {
    int k = j + (j & 8) + 8 * kh;
    a[j] = s[row * 32 + k];
  }
  return a;
}

// Stage an f32 C-tile (8 regs/lane) into a 16x32 f16 LDS tile at column block.
// C layout: reg r <-> row (r + 8*kh), column = lane&15.
__device__ __forceinline__ void stage_C_to_lds(_Float16* s, int colBase,
                                               const float* p8) {
  const int lane = threadIdx.x & 31;
  const int n    = lane & 15;
  const int kh   = lane >> 4;
#pragma unroll
  for (int r = 0; r < 8; ++r)
    s[(r + 8 * kh) * 32 + colBase + n] = (_Float16)p8[r];
}

// ---------------------------------------------------------------------------
// Prep kernels: one-shot f32 -> f16 conversions (move all cvt work out of the
// GEMM inner loops; f16 weights fit in L2 at half the footprint).
// ---------------------------------------------------------------------------
__global__ __launch_bounds__(256) void conv_g(const float* __restrict__ src,
                                              _Float16* __restrict__ dst) {
  int i = (blockIdx.x * 256 + threadIdx.x) * 4;
#pragma unroll
  for (int j = 0; j < 4; ++j) dst[i + j] = (_Float16)src[i + j];
}

// W[c][d] (c=hz rows, d cols) -> W16[c][d] and WT16[d][c]
__global__ __launch_bounds__(256) void conv_w(const float* __restrict__ src,
                                              _Float16* __restrict__ w16,
                                              _Float16* __restrict__ wt16) {
  int i = (blockIdx.x * 256 + threadIdx.x) * 4;
  int c = i / DTOK, d = i % DTOK;
#pragma unroll
  for (int j = 0; j < 4; ++j) {
    _Float16 h = (_Float16)src[i + j];
    w16[i + j] = h;
    wt16[(d + j) * HZ + c] = h;
  }
}

// ---------------------------------------------------------------------------
// Kernel 1: K[n,hz] = g[n,:] . Wk[hz,:]  and same for Q. 16x32 tile per wave
// (A reused across 4 WMMAs). Also writes transposed f16 copies [hz][tok].
// ---------------------------------------------------------------------------
__global__ __launch_bounds__(256) void kq_gemm(
    const _Float16* __restrict__ g16, const _Float16* __restrict__ Wk16,
    const _Float16* __restrict__ Wq16, _Float16* __restrict__ Kb,
    _Float16* __restrict__ Qb, _Float16* __restrict__ KbT,
    _Float16* __restrict__ QbT) {
  const int wave = (blockIdx.x * blockDim.x + threadIdx.x) >> 5;
  const int n0   = (wave >> 5) * 16;  // 128 token-row tiles
  const int hz0  = (wave & 31) * 32;  // 32 hz column pairs
  v8f ck0 = zero8(), ck1 = zero8(), cq0 = zero8(), cq1 = zero8();
  for (int d0 = 0; d0 < DTOK; d0 += 32) {
    v16h a = load_A_f16(g16 + n0 * DTOK + d0, DTOK);
    // B(k=d, n=hz) = W16[(hz0+n)*DTOK + d0 + k] : per-lane contiguous
    v16h bk0 = load_B_f16(Wk16 + (hz0 + 0) * DTOK + d0, 1, DTOK);
    v16h bk1 = load_B_f16(Wk16 + (hz0 + 16) * DTOK + d0, 1, DTOK);
    v16h bq0 = load_B_f16(Wq16 + (hz0 + 0) * DTOK + d0, 1, DTOK);
    v16h bq1 = load_B_f16(Wq16 + (hz0 + 16) * DTOK + d0, 1, DTOK);
    ck0 = wmma16(a, bk0, ck0);
    ck1 = wmma16(a, bk1, ck1);
    cq0 = wmma16(a, bq0, cq0);
    cq1 = wmma16(a, bq1, cq1);
  }
  const int lane = threadIdx.x & 31;
  const int n = lane & 15, kh = lane >> 4;
#pragma unroll
  for (int r = 0; r < 8; ++r) {
    int tok = n0 + r + 8 * kh;
    _Float16 k0v = (_Float16)ck0[r], k1v = (_Float16)ck1[r];
    _Float16 q0v = (_Float16)cq0[r], q1v = (_Float16)cq1[r];
    Kb[tok * HZ + hz0 + n]       = k0v;
    Kb[tok * HZ + hz0 + 16 + n]  = k1v;
    Qb[tok * HZ + hz0 + n]       = q0v;
    Qb[tok * HZ + hz0 + 16 + n]  = q1v;
    KbT[(hz0 + n) * NTOK + tok]      = k0v;
    KbT[(hz0 + 16 + n) * NTOK + tok] = k1v;
    QbT[(hz0 + n) * NTOK + tok]      = q0v;
    QbT[(hz0 + 16 + n) * NTOK + tok] = q1v;
  }
}

// ---------------------------------------------------------------------------
// Kernel 2: softmax stats. For each (h, q): m = max_k beta*<K_k,Q_q>,
//           l = sum_k exp(beta*s - m). One wave per (h, 16-q tile),
//           streaming over 128 k-tiles with WMMA scores (z=64 -> 2 WMMAs).
// ---------------------------------------------------------------------------
__global__ __launch_bounds__(256) void attn_stats(
    const _Float16* __restrict__ Kb, const _Float16* __restrict__ Qb,
    float* __restrict__ mb, float* __restrict__ lb) {
  const int wave = (blockIdx.x * blockDim.x + threadIdx.x) >> 5;
  const int h  = wave >> 7;
  const int q0 = (wave & 127) * 16;
  const int lane = threadIdx.x & 31;
  const int kh = lane >> 4;

  const v16h a0 = load_A_f16(Qb + q0 * HZ + h * ZD + 0,  HZ);
  const v16h a1 = load_A_f16(Qb + q0 * HZ + h * ZD + 32, HZ);

  float mv[8], lv[8];
#pragma unroll
  for (int r = 0; r < 8; ++r) { mv[r] = -INFINITY; lv[r] = 0.f; }

  for (int k0 = 0; k0 < NTOK; k0 += 16) {
    // B(k=z, n=k-token) = Kb[(k0+n)*HZ + h*ZD + z] : per-lane contiguous in z
    v16h b0 = load_B_f16(Kb + k0 * HZ + h * ZD + 0,  1, HZ);
    v16h b1 = load_B_f16(Kb + k0 * HZ + h * ZD + 32, 1, HZ);
    v8f s = wmma16(a1, b1, wmma16(a0, b0, zero8()));
#pragma unroll
    for (int r = 0; r < 8; ++r) {
      float x  = BETA * s[r];
      float mn = fmaxf(mv[r], x);
      lv[r] = lv[r] * __expf(mv[r] - mn) + __expf(x - mn);
      mv[r] = mn;
    }
  }
  // merge (m,l) across the 16 lanes holding one row (width-16 xor shuffles)
#pragma unroll
  for (int off = 1; off < 16; off <<= 1) {
#pragma unroll
    for (int r = 0; r < 8; ++r) {
      float mo = __shfl_xor(mv[r], off, 16);
      float lo = __shfl_xor(lv[r], off, 16);
      float mn = fmaxf(mv[r], mo);
      lv[r] = lv[r] * __expf(mv[r] - mn) + lo * __expf(mo - mn);
      mv[r] = mn;
    }
  }
  if ((lane & 15) == 0) {
#pragma unroll
    for (int r = 0; r < 8; ++r) {
      int q = q0 + r + 8 * kh;
      mb[h * NTOK + q] = mv[r];
      lb[h * NTOK + q] = lv[r];
    }
  }
}

// ---------------------------------------------------------------------------
// Kernel 3: KA[q,h,z] = sum_k softmax_row(q)[k] * K[k,h,z]
//   One wave per (h, q-tile). Per 32-k step: 4 score WMMAs + exp +
//   LDS layout-bounce + 4 value WMMAs (z in 4 groups of 16).
// ---------------------------------------------------------------------------
__global__ __launch_bounds__(256) void attn_t1(
    const _Float16* __restrict__ Kb, const _Float16* __restrict__ Qb,
    const _Float16* __restrict__ KbT, const float* __restrict__ mb,
    const float* __restrict__ lb, _Float16* __restrict__ KA) {
  __shared__ _Float16 Plds[8][16 * 32];
  _Float16* P = Plds[threadIdx.x >> 5];

  const int wave = (blockIdx.x * blockDim.x + threadIdx.x) >> 5;
  const int h  = wave >> 7;
  const int q0 = (wave & 127) * 16;
  const int lane = threadIdx.x & 31;
  const int kh = lane >> 4, n = lane & 15;

  const v16h a0 = load_A_f16(Qb + q0 * HZ + h * ZD + 0,  HZ);
  const v16h a1 = load_A_f16(Qb + q0 * HZ + h * ZD + 32, HZ);

  float mrow[8], linv[8];
#pragma unroll
  for (int r = 0; r < 8; ++r) {
    int q = q0 + r + 8 * kh;
    mrow[r] = mb[h * NTOK + q];
    linv[r] = 1.0f / lb[h * NTOK + q];
  }

  v8f acc[4];
#pragma unroll
  for (int zg = 0; zg < 4; ++zg) acc[zg] = zero8();

  for (int k0 = 0; k0 < NTOK; k0 += 32) {
#pragma unroll
    for (int sub = 0; sub < 2; ++sub) {
      int kk = k0 + sub * 16;
      v16h b0 = load_B_f16(Kb + kk * HZ + h * ZD + 0,  1, HZ);
      v16h b1 = load_B_f16(Kb + kk * HZ + h * ZD + 32, 1, HZ);
      v8f s = wmma16(a1, b1, wmma16(a0, b0, zero8()));
      float p[8];
#pragma unroll
      for (int r = 0; r < 8; ++r)
        p[r] = __expf(BETA * s[r] - mrow[r]) * linv[r];
      stage_C_to_lds(P, sub * 16, p);
    }
    asm volatile("s_wait_dscnt 0" ::: "memory");
    v16h pa = load_A_lds(P);  // P: 16 q-rows x 32 k-cols
#pragma unroll
    for (int zg = 0; zg < 4; ++zg) {
      // B(k=k_local, n=z) = KbT[(h*ZD+zg*16+n)*NTOK + k0 + k] : contiguous
      v16h bz = load_B_f16(KbT + (h * ZD + zg * 16) * NTOK + k0, 1, NTOK);
      acc[zg] = wmma16(pa, bz, acc[zg]);
    }
  }
#pragma unroll
  for (int zg = 0; zg < 4; ++zg)
#pragma unroll
    for (int r = 0; r < 8; ++r)
      KA[(q0 + r + 8 * kh) * HZ + h * ZD + zg * 16 + n] = (_Float16)acc[zg][r];
}

// ---------------------------------------------------------------------------
// Kernel 4: QA[k,h,z] = sum_q softmax_row(q)[k] * Q[q,h,z]   (A2^T . Q)
//   One wave per (h, k-tile). Scores computed transposed (rows=k, cols=q),
//   per-COLUMN softmax scaling using precomputed (m,l), same LDS bounce.
// ---------------------------------------------------------------------------
__global__ __launch_bounds__(256) void attn_t2(
    const _Float16* __restrict__ Kb, const _Float16* __restrict__ Qb,
    const _Float16* __restrict__ QbT, const float* __restrict__ mb,
    const float* __restrict__ lb, _Float16* __restrict__ QA) {
  __shared__ _Float16 Plds[8][16 * 32];
  _Float16* P = Plds[threadIdx.x >> 5];

  const int wave = (blockIdx.x * blockDim.x + threadIdx.x) >> 5;
  const int h  = wave >> 7;
  const int k0 = (wave & 127) * 16;
  const int lane = threadIdx.x & 31;
  const int kh = lane >> 4, n = lane & 15;

  const v16h a0 = load_A_f16(Kb + k0 * HZ + h * ZD + 0,  HZ);
  const v16h a1 = load_A_f16(Kb + k0 * HZ + h * ZD + 32, HZ);

  v8f acc[4];
#pragma unroll
  for (int zg = 0; zg < 4; ++zg) acc[zg] = zero8();

  for (int q0 = 0; q0 < NTOK; q0 += 32) {
#pragma unroll
    for (int sub = 0; sub < 2; ++sub) {
      int qq = q0 + sub * 16;
      // B(k=z, n=q) = Qb[(qq+n)*HZ + h*ZD + z]
      v16h b0 = load_B_f16(Qb + qq * HZ + h * ZD + 0,  1, HZ);
      v16h b1 = load_B_f16(Qb + qq * HZ + h * ZD + 32, 1, HZ);
      v8f s = wmma16(a1, b1, wmma16(a0, b0, zero8()));  // rows=k, cols=q
      int q = qq + n;                   // this lane's column (query index)
      float mq = mb[h * NTOK + q];
      float li = 1.0f / lb[h * NTOK + q];
      float p[8];
#pragma unroll
      for (int r = 0; r < 8; ++r)
        p[r] = __expf(BETA * s[r] - mq) * li;
      stage_C_to_lds(P, sub * 16, p);
    }
    asm volatile("s_wait_dscnt 0" ::: "memory");
    v16h pa = load_A_lds(P);  // P^T: 16 k-rows x 32 q-cols
#pragma unroll
    for (int zg = 0; zg < 4; ++zg) {
      v16h bz = load_B_f16(QbT + (h * ZD + zg * 16) * NTOK + q0, 1, NTOK);
      acc[zg] = wmma16(pa, bz, acc[zg]);
    }
  }
#pragma unroll
  for (int zg = 0; zg < 4; ++zg)
#pragma unroll
    for (int r = 0; r < 8; ++r)
      QA[(k0 + r + 8 * kh) * HZ + h * ZD + zg * 16 + n] = (_Float16)acc[zg][r];
}

// ---------------------------------------------------------------------------
// Kernel 5: out[n,d] = -( KA[n,:] . Wq_flat[:,d]  +  QA[n,:] . Wk_flat[:,d] )
//   B tiles read from transposed f16 weights WT16[d][hz]: per-lane contiguous.
//   16x32 output tile per wave (A reuse across 2 WMMAs per source).
// ---------------------------------------------------------------------------
__global__ __launch_bounds__(256) void final_proj(
    const _Float16* __restrict__ KA, const _Float16* __restrict__ QA,
    const _Float16* __restrict__ WkT, const _Float16* __restrict__ WqT,
    float* __restrict__ out) {
  const int wave = (blockIdx.x * blockDim.x + threadIdx.x) >> 5;
  const int n0 = (wave >> 5) * 16;  // 128 row tiles
  const int d0 = (wave & 31) * 32;  // 32 col pairs
  v8f acc0 = zero8(), acc1 = zero8();
  for (int c0 = 0; c0 < HZ; c0 += 32) {
    v16h a1 = load_A_f16(KA + n0 * HZ + c0, HZ);
    // B(k=c, n=d) = WT[(d0+n)*HZ + c0 + k] : per-lane contiguous
    acc0 = wmma16(a1, load_B_f16(WqT + (d0 + 0) * HZ + c0, 1, HZ), acc0);
    acc1 = wmma16(a1, load_B_f16(WqT + (d0 + 16) * HZ + c0, 1, HZ), acc1);
    v16h a2 = load_A_f16(QA + n0 * HZ + c0, HZ);
    acc0 = wmma16(a2, load_B_f16(WkT + (d0 + 0) * HZ + c0, 1, HZ), acc0);
    acc1 = wmma16(a2, load_B_f16(WkT + (d0 + 16) * HZ + c0, 1, HZ), acc1);
  }
  const int lane = threadIdx.x & 31;
  const int n = lane & 15, kh = lane >> 4;
#pragma unroll
  for (int r = 0; r < 8; ++r) {
    int row = n0 + r + 8 * kh;
    out[row * DTOK + d0 + n]      = -acc0[r];
    out[row * DTOK + d0 + 16 + n] = -acc1[r];
  }
}

// ---------------------------------------------------------------------------
extern "C" void kernel_launch(void* const* d_in, const int* in_sizes, int n_in,
                              void* d_out, int out_size, void* d_ws,
                              size_t ws_size, hipStream_t stream) {
  (void)in_sizes; (void)n_in; (void)out_size; (void)ws_size;
  const float* g  = (const float*)d_in[0];
  const float* Wk = (const float*)d_in[1];
  const float* Wq = (const float*)d_in[2];
  float* out = (float*)d_out;

  // Workspace layout (f16 except stats):
  //   g16[NTOK*DTOK], Wk16/Wq16[HZ*DTOK], WkT/WqT[DTOK*HZ],
  //   Kb/Qb[NTOK*HZ], KbT/QbT[HZ*NTOK], KA/QA[NTOK*HZ], mb/lb f32[NH*NTOK]
  _Float16* g16  = (_Float16*)d_ws;
  _Float16* Wk16 = g16  + (size_t)NTOK * DTOK;
  _Float16* Wq16 = Wk16 + (size_t)HZ * DTOK;
  _Float16* WkT  = Wq16 + (size_t)HZ * DTOK;
  _Float16* WqT  = WkT  + (size_t)DTOK * HZ;
  _Float16* Kb   = WqT  + (size_t)DTOK * HZ;
  _Float16* Qb   = Kb   + (size_t)NTOK * HZ;
  _Float16* KbT  = Qb   + (size_t)NTOK * HZ;
  _Float16* QbT  = KbT  + (size_t)HZ * NTOK;
  _Float16* KA   = QbT  + (size_t)HZ * NTOK;
  _Float16* QA   = KA   + (size_t)NTOK * HZ;
  float*    mb   = (float*)(QA + (size_t)NTOK * HZ);
  float*    lb   = mb + (size_t)NH * NTOK;

  // 0) f32 -> f16 conversions (+ transposed weight copies)
  conv_g<<<(NTOK * DTOK) / 1024, 256, 0, stream>>>(g, g16);
  conv_w<<<(HZ * DTOK) / 1024, 256, 0, stream>>>(Wk, Wk16, WkT);
  conv_w<<<(HZ * DTOK) / 1024, 256, 0, stream>>>(Wq, Wq16, WqT);
  // 1) K/Q projections: 128x32 tiles of 16x32, 8 waves/block
  kq_gemm<<<512, 256, 0, stream>>>(g16, Wk16, Wq16, Kb, Qb, KbT, QbT);
  // 2) softmax stats per (h,q): 2048 waves
  attn_stats<<<256, 256, 0, stream>>>(Kb, Qb, mb, lb);
  // 3) KA = A2 . K : 2048 waves
  attn_t1<<<256, 256, 0, stream>>>(Kb, Qb, KbT, mb, lb, KA);
  // 4) QA = A2^T . Q : 2048 waves
  attn_t2<<<256, 256, 0, stream>>>(Kb, Qb, QbT, mb, lb, QA);
  // 5) out = -(KA.Wq + QA.Wk): 4096 tiles of 16x32
  final_proj<<<512, 256, 0, stream>>>(KA, QA, WkT, WqT, out);
}
